// TransformerEncoderLayer_83038897701191
// MI455X (gfx1250) — compile-verified, matrix operand order checked by reading, not run on hardware
//
#include <hip/hip_runtime.h>
#include <cstddef>

// ---------------------------------------------------------------------------
// Types for CDNA5 WMMA (wave32, 16x16x32 bf16 -> f32)
// ---------------------------------------------------------------------------
typedef __bf16 bf16_t;
typedef __attribute__((ext_vector_type(16))) __bf16 v16bf;
typedef __attribute__((ext_vector_type(8)))  float  v8f;

__device__ __forceinline__ v8f wmma_bf16(v16bf a, v16bf b, v8f c) {
    return __builtin_amdgcn_wmma_f32_16x16x32_bf16(
        /*neg_a=*/false, a, /*neg_b=*/false, b,
        /*c_mod=*/(short)0, c, /*reuse_a=*/false, /*reuse_b=*/false);
}

// Load one 16x32 (MxK) bf16 fragment from LDS (row-major, rowStride in elems).
// Per ISA 7.12.2: lanes 0-15 hold row r=lane, K = {0..7, 16..23};
// lanes 16-31 hold row r=lane-16, K = {8..15, 24..31}.
// The same loader serves the B-matrix (KxN) when the source is stored
// row-major as N x K (rows = output columns).
__device__ __forceinline__ v16bf load_frag(const bf16_t* lds, int rowStride,
                                           int row0, int k0, int lane) {
    int r   = lane & 15;
    int sel = lane >> 4;
    const bf16_t* p = lds + (size_t)(row0 + r) * rowStride + k0;
    union { v16bf v; uint4 q[2]; } u;
    u.q[0] = *(const uint4*)(p + sel * 8);
    u.q[1] = *(const uint4*)(p + 16 + sel * 8);
    return u.v;
}

__device__ __forceinline__ v8f v8f_zero() {
    v8f z = {0.f, 0.f, 0.f, 0.f, 0.f, 0.f, 0.f, 0.f};
    return z;
}

// ---------------------------------------------------------------------------
// GEMM: C[M x N] = A[M x K] @ W[N x K]^T + bias  (optional ReLU)
// fp32 in/out, bf16 WMMA compute.
// Block tile 128x128, 8 waves (4 M x 2 N), each wave 32x64 = 2x4 WMMA tiles.
// K-step 32, register double-buffered global->LDS staging.
// LDS rows padded to 40 bf16 (80B, 16B-aligned) -> conflict-free b128 reads.
// ---------------------------------------------------------------------------
template <bool RELU>
__global__ __launch_bounds__(256) void gemm_bias_kernel(
    const float* __restrict__ A, const float* __restrict__ W,
    const float* __restrict__ bias, float* __restrict__ C,
    int M, int N, int K) {
    constexpr int BM = 128, BN = 128, KB = 32, LDT = 40;
    __shared__ __align__(16) bf16_t As[BM * LDT];
    __shared__ __align__(16) bf16_t Ws[BN * LDT];

    const int tid   = threadIdx.x;
    const int lane  = tid & 31;
    const int wave  = tid >> 5;
    const int waveM = wave >> 1;   // 0..3 -> 32-row strip
    const int waveN = wave & 1;    // 0..1 -> 64-col strip
    const int blockM = blockIdx.y * BM;
    const int blockN = blockIdx.x * BN;

    v8f acc[2][4];
    #pragma unroll
    for (int i = 0; i < 2; ++i)
        #pragma unroll
        for (int j = 0; j < 4; ++j) acc[i][j] = v8f_zero();

    // each thread stages 4 float4 of A and 4 float4 of W per K-step
    float4 ra[4], rw[4];
    auto loadG = [&](const float* __restrict__ P, int k0, float4* r) {
        #pragma unroll
        for (int i = 0; i < 4; ++i) {
            int idx  = tid + i * 256;          // 0..1023 float4 slots
            int row  = idx >> 3;               // 8 float4 per 32-col row
            int col4 = (idx & 7) * 4;
            r[i] = *(const float4*)(P + (size_t)row * K + k0 + col4);
        }
    };
    auto storeL = [&](bf16_t* __restrict__ S, const float4* r) {
        #pragma unroll
        for (int i = 0; i < 4; ++i) {
            int idx  = tid + i * 256;
            int row  = idx >> 3;
            int col4 = (idx & 7) * 4;
            bf16_t* d = S + row * LDT + col4;
            d[0] = (bf16_t)r[i].x; d[1] = (bf16_t)r[i].y;
            d[2] = (bf16_t)r[i].z; d[3] = (bf16_t)r[i].w;
        }
    };

    const float* Ab = A + (size_t)blockM * K;
    const float* Wb = W + (size_t)blockN * K;
    const int kTiles = K / KB;

    loadG(Ab, 0, ra);
    loadG(Wb, 0, rw);

    for (int kt = 0; kt < kTiles; ++kt) {
        storeL(As, ra);
        storeL(Ws, rw);
        __syncthreads();

        if (kt + 1 < kTiles) {                 // prefetch next tile into VGPRs
            loadG(Ab, (kt + 1) * KB, ra);
            loadG(Wb, (kt + 1) * KB, rw);
        }

        v16bf a0 = load_frag(As, LDT, waveM * 32,      0, lane);
        v16bf a1 = load_frag(As, LDT, waveM * 32 + 16, 0, lane);
        #pragma unroll
        for (int t = 0; t < 4; ++t) {
            v16bf b = load_frag(Ws, LDT, waveN * 64 + t * 16, 0, lane);
            acc[0][t] = wmma_bf16(a0, b, acc[0][t]);
            acc[1][t] = wmma_bf16(a1, b, acc[1][t]);
        }
        __syncthreads();
    }

    // Store: C/D layout -> lane&15 = N, lane>>4 selects rows {0..7}/{8..15}
    const int nl  = lane & 15;
    const int sel = lane >> 4;
    #pragma unroll
    for (int tm = 0; tm < 2; ++tm) {
        #pragma unroll
        for (int tn = 0; tn < 4; ++tn) {
            int n = blockN + waveN * 64 + tn * 16 + nl;
            float bv = bias[n];
            int mbase = blockM + waveM * 32 + tm * 16 + sel * 8;
            #pragma unroll
            for (int v = 0; v < 8; ++v) {
                float o = acc[tm][tn][v] + bv;
                if (RELU) o = fmaxf(o, 0.f);
                C[(size_t)(mbase + v) * N + n] = o;
            }
        }
    }
}

// ---------------------------------------------------------------------------
// Flash attention: one block per (b, h, Q-row-block of 64). Br=Bc=64, hd=64.
// qkv row layout (per reference reshape): per head h, cols [h*192 .. h*192+191]
// are [q(64) | k(64) | v(64)].  Output written as vals[b, l, h*64 + d].
// bf16 LDS tiles padded to stride 72 (144B) -> conflict-free b128 reads.
// ---------------------------------------------------------------------------
__global__ __launch_bounds__(256) void flash_attn_kernel(
    const float* __restrict__ qkv, const int* __restrict__ mask,
    float* __restrict__ vals) {
    constexpr int Br = 64, Bc = 64, HD = 64, LL = 2048, DD = 512, QW = 1536;
    constexpr int LQ = 72;   // padded bf16 row stride

    const int bh = blockIdx.y;       // 0..31
    const int b  = bh >> 3;
    const int h  = bh & 7;
    const int q0 = blockIdx.x * Br;

    __shared__ __align__(16) bf16_t Qs[Br * LQ];
    __shared__ __align__(16) bf16_t Ks[Bc * LQ];
    __shared__ __align__(16) bf16_t Vt[HD * LQ];   // transposed: [d][s]
    __shared__ __align__(16) float  Ss[Br * Bc];
    __shared__ __align__(16) bf16_t Ps[Br * LQ];
    __shared__ float mrow[Br], lrow[Br], arow[Br];
    __shared__ float red[Br * 4];

    const int tid   = threadIdx.x;
    const int lane  = tid & 31;
    const int wave  = tid >> 5;
    const int waveM = wave >> 1;     // 0..3 -> M tile
    const int waveN = wave & 1;      // 0..1 -> N half (2 tiles each)

    const float* qbase = qkv + (size_t)b * LL * QW + (size_t)h * 192;

    // Load Q block, pre-scaled by 1/sqrt(64) = 0.125
    {
        int row = tid >> 2, cg = (tid & 3) * 16;
        const float* src = qbase + (size_t)(q0 + row) * QW + cg;
        bf16_t* dst = Qs + row * LQ + cg;
        #pragma unroll
        for (int i = 0; i < 16; i += 4) {
            float4 f = *(const float4*)(src + i);
            dst[i + 0] = (bf16_t)(f.x * 0.125f);
            dst[i + 1] = (bf16_t)(f.y * 0.125f);
            dst[i + 2] = (bf16_t)(f.z * 0.125f);
            dst[i + 3] = (bf16_t)(f.w * 0.125f);
        }
    }
    if (tid < Br) { mrow[tid] = -1e30f; lrow[tid] = 0.f; }

    v8f o[2];
    o[0] = v8f_zero(); o[1] = v8f_zero();
    __syncthreads();

    for (int s0 = 0; s0 < LL; s0 += Bc) {
        // Stage K (row-major s x d) and V (transposed d x s)
        {
            int row = tid >> 2, cg = (tid & 3) * 16;
            const float* ksrc = qbase + (size_t)(s0 + row) * QW + 64 + cg;
            const float* vsrc = qbase + (size_t)(s0 + row) * QW + 128 + cg;
            bf16_t* kd = Ks + row * LQ + cg;
            #pragma unroll
            for (int i = 0; i < 16; i += 4) {
                float4 f = *(const float4*)(ksrc + i);
                kd[i + 0] = (bf16_t)f.x; kd[i + 1] = (bf16_t)f.y;
                kd[i + 2] = (bf16_t)f.z; kd[i + 3] = (bf16_t)f.w;
            }
            #pragma unroll
            for (int i = 0; i < 16; ++i)
                Vt[(cg + i) * LQ + row] = (bf16_t)vsrc[i];
        }
        __syncthreads();

        // S = Qs @ Ks^T  (each wave: 1 m-tile x 2 n-tiles, K=64 -> 2 steps)
        #pragma unroll
        for (int tn = 0; tn < 2; ++tn) {
            v8f s = v8f_zero();
            #pragma unroll
            for (int kk = 0; kk < 2; ++kk) {
                v16bf a  = load_frag(Qs, LQ, waveM * 16,           kk * 32, lane);
                v16bf bb = load_frag(Ks, LQ, waveN * 32 + tn * 16, kk * 32, lane);
                s = wmma_bf16(a, bb, s);
            }
            int n  = waveN * 32 + tn * 16 + (lane & 15);
            int mb = waveM * 16 + (lane >> 4) * 8;
            #pragma unroll
            for (int v = 0; v < 8; ++v) Ss[(mb + v) * Bc + n] = s[v];
        }
        __syncthreads();

        // Online softmax: 4 threads per row, 16 cols each
        {
            int row = tid >> 2, part = tid & 3, cg = part * 16;
            const int* mp = mask + (size_t)(q0 + row) * LL + s0 + cg;
            float* sp = Ss + row * Bc + cg;
            float sv[16];
            float mx = -1e30f;
            #pragma unroll
            for (int i = 0; i < 16; ++i) {
                float s = sp[i];
                if (mp[i] == 0) s = -1e30f;
                sv[i] = s;
                mx = fmaxf(mx, s);
            }
            red[row * 4 + part] = mx;
            __syncthreads();
            float bmax = fmaxf(fmaxf(red[row * 4 + 0], red[row * 4 + 1]),
                               fmaxf(red[row * 4 + 2], red[row * 4 + 3]));
            float mold = mrow[row];
            float mnew = fmaxf(mold, bmax);
            float alpha = __expf(mold - mnew);
            float psum = 0.f;
            bf16_t* pd = Ps + row * LQ + cg;
            #pragma unroll
            for (int i = 0; i < 16; ++i) {
                float p = __expf(sv[i] - mnew);
                psum += p;
                pd[i] = (bf16_t)p;
            }
            red[row * 4 + part] = psum;   // same-wave lockstep after reads
            __syncthreads();
            if (part == 0) {
                float bsum = red[row * 4 + 0] + red[row * 4 + 1] +
                             red[row * 4 + 2] + red[row * 4 + 3];
                lrow[row] = lrow[row] * alpha + bsum;
                mrow[row] = mnew;
                arow[row] = alpha;
            }
        }
        __syncthreads();

        // Rescale O by alpha and accumulate O += P @ V
        {
            int mb = waveM * 16 + (lane >> 4) * 8;
            #pragma unroll
            for (int tn = 0; tn < 2; ++tn)
                #pragma unroll
                for (int v = 0; v < 8; ++v) o[tn][v] *= arow[mb + v];
            #pragma unroll
            for (int kk = 0; kk < 2; ++kk) {
                v16bf a = load_frag(Ps, LQ, waveM * 16, kk * 32, lane);
                #pragma unroll
                for (int tn = 0; tn < 2; ++tn) {
                    v16bf bb = load_frag(Vt, LQ, waveN * 32 + tn * 16, kk * 32, lane);
                    o[tn] = wmma_bf16(a, bb, o[tn]);
                }
            }
        }
        __syncthreads();
    }

    // Finalize: O /= l, scatter to vals[b, q0+row, h*64 + d]
    {
        int mb = waveM * 16 + (lane >> 4) * 8;
        int nl = lane & 15;
        #pragma unroll
        for (int tn = 0; tn < 2; ++tn) {
            int d = waveN * 32 + tn * 16 + nl;
            #pragma unroll
            for (int v = 0; v < 8; ++v) {
                int row = mb + v;
                float inv = 1.f / lrow[row];
                vals[(size_t)(b * LL + q0 + row) * DD + h * HD + d] = o[tn][v] * inv;
            }
        }
    }
}

// ---------------------------------------------------------------------------
// Fused residual add + LayerNorm over D=512 (one block per row, 256 threads)
// out = ((x+y) - mean) * rsqrt(var + eps) * g + beta
// ---------------------------------------------------------------------------
__global__ __launch_bounds__(256) void add_ln_kernel(
    const float* __restrict__ x, const float* __restrict__ y,
    const float* __restrict__ g, const float* __restrict__ beta,
    float* __restrict__ out) {
    constexpr int D = 512;
    const int row = blockIdx.x;
    const int tid = threadIdx.x;
    __shared__ float red[256];

    const float* xr = x + (size_t)row * D;
    const float* yr = y + (size_t)row * D;
    float v0 = xr[tid] + yr[tid];
    float v1 = xr[tid + 256] + yr[tid + 256];

    red[tid] = v0 + v1;
    __syncthreads();
    for (int off = 128; off > 0; off >>= 1) {
        if (tid < off) red[tid] += red[tid + off];
        __syncthreads();
    }
    float mean = red[0] * (1.f / D);
    __syncthreads();

    float d0 = v0 - mean, d1 = v1 - mean;
    red[tid] = d0 * d0 + d1 * d1;
    __syncthreads();
    for (int off = 128; off > 0; off >>= 1) {
        if (tid < off) red[tid] += red[tid + off];
        __syncthreads();
    }
    float inv = rsqrtf(red[0] * (1.f / D) + 1e-5f);

    float* orow = out + (size_t)row * D;
    orow[tid]       = d0 * inv * g[tid]       + beta[tid];
    orow[tid + 256] = d1 * inv * g[tid + 256] + beta[tid + 256];
}

// ---------------------------------------------------------------------------
// Launch
// ---------------------------------------------------------------------------
extern "C" void kernel_launch(void* const* d_in, const int* in_sizes, int n_in,
                              void* d_out, int out_size, void* d_ws, size_t ws_size,
                              hipStream_t stream) {
    constexpr int Bt = 4, L = 2048, D = 512, DFF = 2048;
    constexpr int M = Bt * L;  // 8192

    const float* src   = (const float*)d_in[0];
    const int*   mask  = (const int*)d_in[1];
    const float* qkv_w = (const float*)d_in[2];
    const float* qkv_b = (const float*)d_in[3];
    const float* out_w = (const float*)d_in[4];
    const float* out_b = (const float*)d_in[5];
    const float* w1    = (const float*)d_in[6];
    const float* b1    = (const float*)d_in[7];
    const float* w2    = (const float*)d_in[8];
    const float* b2    = (const float*)d_in[9];
    const float* g1    = (const float*)d_in[10];
    const float* be1   = (const float*)d_in[11];
    const float* g2    = (const float*)d_in[12];
    const float* be2   = (const float*)d_in[13];
    float* out = (float*)d_out;

    float* ws   = (float*)d_ws;
    float* qkv  = ws;                            // 8192 x 1536
    float* valb = qkv  + (size_t)M * 3 * D;      // 8192 x 512
    float* sa   = valb + (size_t)M * D;          // 8192 x 512
    float* x1   = sa   + (size_t)M * D;          // 8192 x 512
    float* hbuf = x1   + (size_t)M * D;          // 8192 x 2048
    float* ff   = hbuf + (size_t)M * DFF;        // 8192 x 512

    const dim3 blk(256);

    // 1) QKV projection
    gemm_bias_kernel<false><<<dim3((3 * D) / 128, M / 128), blk, 0, stream>>>(
        src, qkv_w, qkv_b, qkv, M, 3 * D, D);
    // 2) attention
    flash_attn_kernel<<<dim3(L / 64, Bt * 8), blk, 0, stream>>>(qkv, mask, valb);
    // 3) output projection
    gemm_bias_kernel<false><<<dim3(D / 128, M / 128), blk, 0, stream>>>(
        valb, out_w, out_b, sa, M, D, D);
    // 4) residual + LN1
    add_ln_kernel<<<dim3(M), blk, 0, stream>>>(src, sa, g1, be1, x1);
    // 5) FFN up + ReLU
    gemm_bias_kernel<true><<<dim3(DFF / 128, M / 128), blk, 0, stream>>>(
        x1, w1, b1, hbuf, M, DFF, D);
    // 6) FFN down
    gemm_bias_kernel<false><<<dim3(D / 128, M / 128), blk, 0, stream>>>(
        hbuf, w2, b2, ff, M, D, DFF);
    // 7) residual + LN2 -> out
    add_ln_kernel<<<dim3(M), blk, 0, stream>>>(x1, ff, g2, be2, out);
}
